// Block_71554155151851
// MI455X (gfx1250) — matrix operand order repaired and checked by reading, not compiled, not run on hardware
//
#include <hip/hip_runtime.h>
#include <hip/hip_bf16.h>

typedef __bf16 bf16_t;
typedef bf16_t v16bf __attribute__((ext_vector_type(16)));
typedef float  v8f   __attribute__((ext_vector_type(8)));

#if __has_builtin(__builtin_amdgcn_tensor_load_to_lds) && \
    __has_builtin(__builtin_amdgcn_s_wait_tensorcnt)
#define USE_TDM 1
#else
#define USE_TDM 0
#endif

// padded LDS row strides (avoid 16-way bank conflicts on A-fragment reads)
#define LDF 66    // f32 tiles: 64 cols + 2 pad dwords
#define LDS17 17  // edge_sh: 16 cols + 1 pad dword
#define LDB 520   // bf16 tile: 512 cols + 8 pad bf16 (4 pad dwords)

__device__ __forceinline__ v8f wmma_bf16(v16bf a, v16bf b, v8f c) {
  return __builtin_amdgcn_wmma_f32_16x16x32_bf16(false, a, false, b, (short)0, c,
                                                 false, false);
}

// single v_rcp_f32 (avoid IEEE div_scale/fma expansion in hot paths)
__device__ __forceinline__ float fast_rcp(float x) {
  return __builtin_amdgcn_rcpf(x);
}

__device__ __forceinline__ float silu(float x) {
  return x * fast_rcp(1.f + __expf(-x));
}

#if USE_TDM
typedef unsigned tdm_u32x4 __attribute__((ext_vector_type(4)));
typedef int      tdm_i32x8 __attribute__((ext_vector_type(8)));
typedef int      tdm_i32x4 __attribute__((ext_vector_type(4)));

// Issue one 2D TDM tile load: rows x cols elements, row stride strideElems,
// element size 2^dsizeCode bytes, optional LDS row padding via TDM pad unit.
// D# layout per CDNA5 ISA 8.3/8.4 (group0: count/lds_addr/global_addr/type,
// group1: data_size, pad, tensor dims, tile dims, dim0 stride).
__device__ __forceinline__ void tdm_load_tile_2d(const void* gbase, unsigned ldsAddr,
                                                 unsigned rows, unsigned cols,
                                                 unsigned strideElems, unsigned dsizeCode,
                                                 unsigned padEn, unsigned padIval,
                                                 unsigned padAmt) {
  unsigned long long ga = (unsigned long long)gbase;
  tdm_u32x4 g0;
  g0[0] = 1u;                                   // count=1 (valid user descriptor)
  g0[1] = ldsAddr;                              // lds_addr (bytes)
  g0[2] = (unsigned)ga;                         // global_addr[31:0]
  g0[3] = (unsigned)((ga >> 32) & 0x01FFFFFFull) | (2u << 30);  // addr[56:32], type=2
  tdm_i32x8 g1;
  g1[0] = (int)((dsizeCode << 16) | (padEn << 20) | (padIval << 22) | (padAmt << 25));
  g1[1] = (int)((cols & 0xFFFFu) << 16);                         // tensor_dim0 lo
  g1[2] = (int)(((cols >> 16) & 0xFFFFu) | ((rows & 0xFFFFu) << 16)); // dim0 hi | dim1 lo
  g1[3] = (int)(((rows >> 16) & 0xFFFFu) | ((cols & 0xFFFFu) << 16)); // dim1 hi | tile_dim0
  g1[4] = (int)(rows & 0xFFFFu);                                 // tile_dim1
  g1[5] = (int)strideElems;                                      // tensor_dim0_stride lo
  g1[6] = 0;
  g1[7] = 0;
  tdm_i32x4 gz;
  gz[0] = gz[1] = gz[2] = gz[3] = 0;
  tdm_i32x8 gz8;
  gz8[0] = gz8[1] = gz8[2] = gz8[3] = gz8[4] = gz8[5] = gz8[6] = gz8[7] = 0;
  __builtin_amdgcn_tensor_load_to_lds(g0, g1, gz, gz, gz8, 0);
}
#endif

// ---- B fragment: pre-packed layout, 16 bf16 contiguous per (tile, lane) ----
__device__ __forceinline__ v16bf load_bfrag(const bf16_t* __restrict__ pack,
                                            int ntiles, int kt, int nt, int lane) {
  const bf16_t* p = pack + (((size_t)(kt * ntiles + nt)) * 32 + lane) * 16;
  return *(const v16bf*)p;
}

// ---- A fragment (16x32 bf16) from f32 LDS matrix, per ISA 7.12.2 layout ----
__device__ __forceinline__ v16bf buildA_f32(const float* src, int ld, int m,
                                            int kbase, int hf) {
  const float* r = src + m * ld + kbase + hf * 8;
  v16bf a;
#pragma unroll
  for (int v = 0; v < 4; ++v) {
    a[2 * v]     = (bf16_t)r[2 * v];
    a[2 * v + 1] = (bf16_t)r[2 * v + 1];
    a[8 + 2 * v] = (bf16_t)r[16 + 2 * v];
    a[9 + 2 * v] = (bf16_t)r[17 + 2 * v];
  }
  return a;
}

__device__ __forceinline__ v16bf buildA_bf16(const bf16_t* src, int ld, int m,
                                             int kbase, int hf) {
  const bf16_t* r = src + m * ld + kbase + hf * 8;
  v16bf a;
#pragma unroll
  for (int v = 0; v < 4; ++v) {
    a[2 * v]     = r[2 * v];
    a[2 * v + 1] = r[2 * v + 1];
    a[8 + 2 * v] = r[16 + 2 * v];
    a[9 + 2 * v] = r[17 + 2 * v];
  }
  return a;
}

// A fragment of z = edge_in (x) edge_sh built on the fly.
__device__ __forceinline__ v16bf buildZ(const float* einRow, int kc,
                                        const float* shv) {
  float e0v = einRow[2 * kc], e1v = einRow[2 * kc + 1];
  v16bf a;
#pragma unroll
  for (int v = 0; v < 4; ++v) {
    a[2 * v]     = (bf16_t)(e0v * shv[2 * v]);
    a[2 * v + 1] = (bf16_t)(e0v * shv[2 * v + 1]);
    a[8 + 2 * v] = (bf16_t)(e1v * shv[2 * v]);
    a[9 + 2 * v] = (bf16_t)(e1v * shv[2 * v + 1]);
  }
  return a;
}

// =============== weight packing: row-major f32 [K, srcN] -> B fragments ======
__global__ void pack_b_kernel(const float* __restrict__ W, bf16_t* __restrict__ dst,
                              int K, int N, int srcN) {
  int idx = blockIdx.x * blockDim.x + threadIdx.x;
  int ntiles = N >> 4;
  int total = (K >> 5) * ntiles * 32;
  if (idx >= total) return;
  int lane = idx & 31;
  int tile = idx >> 5;
  int nt = tile % ntiles;
  int kt = tile / ntiles;
  int n  = nt * 16 + (lane & 15);
  int kb = kt * 32 + (lane >> 4) * 8;
  bf16_t* o = dst + (size_t)idx * 16;
#pragma unroll
  for (int v = 0; v < 8; ++v) {
    int k0 = kb + 2 * v + (v >= 4 ? 8 : 0);
    float x0 = (n < srcN) ? W[(size_t)k0 * srcN + n] : 0.f;
    float x1 = (n < srcN) ? W[(size_t)(k0 + 1) * srcN + n] : 0.f;
    o[2 * v]     = (bf16_t)x0;
    o[2 * v + 1] = (bf16_t)x1;
  }
}

// =============== K1: value[E,512] = ((ein (x) sh) @ W_tp) * radialMLP ========
__global__ void __launch_bounds__(256)
k1_value(const float* __restrict__ edge_in, const float* __restrict__ edge_sh,
         const float* __restrict__ emb, const bf16_t* __restrict__ pWtp,
         const bf16_t* __restrict__ pR1, const bf16_t* __restrict__ pR2,
         const float* __restrict__ rb1, const float* __restrict__ rb2,
         bf16_t* __restrict__ value) {
  __shared__ alignas(16) float einL[16 * LDF];
  __shared__ alignas(16) float eshL[16 * LDS17];
  __shared__ alignas(16) float embL[16 * LDF];
  __shared__ alignas(16) float h1L[16 * LDF];
  const int e0 = blockIdx.x * 16;
  const int tid = threadIdx.x;
  const float* einG = edge_in + (size_t)e0 * 64;
  const float* embG = emb + (size_t)e0 * 64;
  const float* eshG = edge_sh + (size_t)e0 * 16;
#if USE_TDM
  if (tid < 32) {
    // f32 tiles 16x64, pad 2 dwords every 64 dwords -> ld 66
    tdm_load_tile_2d(einG, (unsigned)(unsigned long long)(void*)einL, 16, 64, 64, 2, 1, 5, 1);
    tdm_load_tile_2d(embG, (unsigned)(unsigned long long)(void*)embL, 16, 64, 64, 2, 1, 5, 1);
    // f32 tile 16x16, pad 1 dword every 16 dwords -> ld 17
    tdm_load_tile_2d(eshG, (unsigned)(unsigned long long)(void*)eshL, 16, 16, 16, 2, 1, 3, 0);
    __builtin_amdgcn_s_wait_tensorcnt(0);
  }
#else
  for (int i = tid; i < 1024; i += 256) {
    int r = i >> 6, c = i & 63;
    einL[r * LDF + c] = einG[i];
    embL[r * LDF + c] = embG[i];
  }
  { int r = tid >> 4, c = tid & 15; if (tid < 256) eshL[r * LDS17 + c] = eshG[tid]; }
#endif
  __syncthreads();
  const int w = tid >> 5, lane = tid & 31, m = lane & 15, hf = lane >> 4;
  float shv[8];
#pragma unroll
  for (int j = 0; j < 8; ++j) shv[j] = eshL[m * LDS17 + hf * 8 + j];

  // radial layer1 (N=64 -> waves 0..3), silu, stage to LDS
  if (w < 4) {
    v8f r1 = {0.f, 0.f, 0.f, 0.f, 0.f, 0.f, 0.f, 0.f};
#pragma unroll
    for (int kc = 0; kc < 2; ++kc) {
      v16bf a = buildA_f32(embL, LDF, m, kc * 32, hf);
      v16bf b = load_bfrag(pR1, 4, kc, w, lane);
      r1 = wmma_bf16(a, b, r1);
    }
    int n = w * 16 + m;
    float bv = rb1[n];
#pragma unroll
    for (int i = 0; i < 8; ++i) h1L[(i + hf * 8) * LDF + n] = silu(r1[i] + bv);
  }
  __syncthreads();

  v8f zero = {0.f, 0.f, 0.f, 0.f, 0.f, 0.f, 0.f, 0.f};
  v8f acc[4], rc[4];
#pragma unroll
  for (int j = 0; j < 4; ++j) { acc[j] = zero; rc[j] = zero; }

  // main GEMM: M=16, N-slice of 64 per wave, K=1024
  const float* einRow = einL + m * LDF;
#pragma unroll 4
  for (int kc = 0; kc < 32; ++kc) {
    v16bf a = buildZ(einRow, kc, shv);
#pragma unroll
    for (int j = 0; j < 4; ++j) {
      v16bf b = load_bfrag(pWtp, 32, kc, w * 4 + j, lane);
      acc[j] = wmma_bf16(a, b, acc[j]);
    }
  }
  // radial layer2: h1 @ W2, same N columns as main GEMM slice
#pragma unroll
  for (int kc = 0; kc < 2; ++kc) {
    v16bf a = buildA_f32(h1L, LDF, m, kc * 32, hf);
#pragma unroll
    for (int j = 0; j < 4; ++j) {
      v16bf b = load_bfrag(pR2, 32, kc, w * 4 + j, lane);
      rc[j] = wmma_bf16(a, b, rc[j]);
    }
  }
#pragma unroll
  for (int j = 0; j < 4; ++j) {
    int n = w * 64 + j * 16 + m;
    float bv = rb2[n];
#pragma unroll
    for (int i = 0; i < 8; ++i) {
      float val = acc[j][i] * (rc[j][i] + bv);
      value[(size_t)(e0 + i + hf * 8) * 512 + n] = (bf16_t)val;
    }
  }
}

// =============== K2: alpha MLPs (two branches), LN + silu ====================
__device__ void ln_silu_inplace(float* buf, const float* g, const float* be, int tid) {
  if (tid < 32) {
    int br = tid >> 4, m = tid & 15;
    float* r = buf + (br * 16 + m) * LDF;
    float mu = 0.f;
    for (int n = 0; n < 64; ++n) mu += r[n];
    mu *= (1.f / 64.f);
    float var = 0.f;
    for (int n = 0; n < 64; ++n) { float d = r[n] - mu; var += d * d; }
    var *= (1.f / 64.f);
    float rs = rsqrtf(var + 1e-6f);
    const float* gb = g + br * 64;
    const float* bb = be + br * 64;
    for (int n = 0; n < 64; ++n) {
      float x = (r[n] - mu) * rs * gb[n] + bb[n];
      r[n] = silu(x);
    }
  }
}

__global__ void __launch_bounds__(256)
k2_alpha(const float* __restrict__ temb2, const float* __restrict__ temb3,
         const bf16_t* __restrict__ pA1, const bf16_t* __restrict__ pA2,
         const bf16_t* __restrict__ pA3, const float* __restrict__ ab1,
         const float* __restrict__ ag1, const float* __restrict__ abe1,
         const float* __restrict__ ab2, const float* __restrict__ ag2,
         const float* __restrict__ abe2, const float* __restrict__ ab3,
         float* __restrict__ alpha_out) {
  __shared__ alignas(16) float xb[2 * 16 * LDF];
  __shared__ alignas(16) float hb[2 * 16 * LDF];
  __shared__ alignas(16) float ob[2 * 16 * 16];
  const int t0 = blockIdx.x * 16, tid = threadIdx.x;
#if USE_TDM
  if (tid < 32) {
    tdm_load_tile_2d(temb2 + (size_t)t0 * 64, (unsigned)(unsigned long long)(void*)xb,
                     16, 64, 64, 2, 1, 5, 1);
    tdm_load_tile_2d(temb3 + (size_t)t0 * 64,
                     (unsigned)(unsigned long long)(void*)(xb + 16 * LDF),
                     16, 64, 64, 2, 1, 5, 1);
    __builtin_amdgcn_s_wait_tensorcnt(0);
  }
#else
  for (int i = tid; i < 1024; i += 256) {
    int r = i >> 6, c = i & 63;
    xb[r * LDF + c]            = temb2[(size_t)t0 * 64 + i];
    xb[16 * LDF + r * LDF + c] = temb3[(size_t)t0 * 64 + i];
  }
#endif
  __syncthreads();
  const int w = tid >> 5, lane = tid & 31, m = lane & 15, hf = lane >> 4;
  const int br = w >> 2, nt = w & 3;
  const int bro = br * 16 * LDF;
  v8f zero = {0.f, 0.f, 0.f, 0.f, 0.f, 0.f, 0.f, 0.f};

  {  // layer1: xb -> hb
    v8f c = zero;
#pragma unroll
    for (int kc = 0; kc < 2; ++kc)
      c = wmma_bf16(buildA_f32(xb + bro, LDF, m, kc * 32, hf),
                    load_bfrag(pA1 + br * 4096, 4, kc, nt, lane), c);
    int n = nt * 16 + m;
    float bv = ab1[br * 64 + n];
#pragma unroll
    for (int i = 0; i < 8; ++i) hb[bro + (i + hf * 8) * LDF + n] = c[i] + bv;
  }
  __syncthreads();
  ln_silu_inplace(hb, ag1, abe1, tid);
  __syncthreads();
  {  // layer2: hb -> xb
    v8f c = zero;
#pragma unroll
    for (int kc = 0; kc < 2; ++kc)
      c = wmma_bf16(buildA_f32(hb + bro, LDF, m, kc * 32, hf),
                    load_bfrag(pA2 + br * 4096, 4, kc, nt, lane), c);
    int n = nt * 16 + m;
    float bv = ab2[br * 64 + n];
#pragma unroll
    for (int i = 0; i < 8; ++i) xb[bro + (i + hf * 8) * LDF + n] = c[i] + bv;
  }
  __syncthreads();
  ln_silu_inplace(xb, ag2, abe2, tid);
  __syncthreads();
  if (nt == 0) {  // layer3: xb -> ob (N=16, cols 8..15 zero-padded weights)
    v8f c = zero;
#pragma unroll
    for (int kc = 0; kc < 2; ++kc)
      c = wmma_bf16(buildA_f32(xb + bro, LDF, m, kc * 32, hf),
                    load_bfrag(pA3 + br * 1024, 1, kc, 0, lane), c);
    int n = m;
    float bv = (n < 8) ? ab3[br * 8 + n] : 0.f;
#pragma unroll
    for (int i = 0; i < 8; ++i) ob[br * 256 + (i + hf * 8) * 16 + n] = c[i] + bv;
  }
  __syncthreads();
  if (tid < 128) {
    int mm = tid >> 3, h = tid & 7;
    alpha_out[((size_t)(t0 + mm)) * 8 + h] = ob[mm * 16 + h] * ob[256 + mm * 16 + h];
  }
}

// =============== segment softmax helpers =====================================
__device__ __forceinline__ unsigned fenc(float f) {
  unsigned u = __float_as_uint(f);
  return (u & 0x80000000u) ? ~u : (u | 0x80000000u);
}
__device__ __forceinline__ float fdec(unsigned u) {
  return (u & 0x80000000u) ? __uint_as_float(u & 0x7FFFFFFFu) : __uint_as_float(~u);
}

__global__ void k3_max(const float* __restrict__ alpha, const int* __restrict__ idx0,
                       unsigned* __restrict__ segmax, int n) {
  int i = blockIdx.x * 256 + threadIdx.x;
  if (i >= n) return;
  int t = i >> 3, h = i & 7;
  atomicMax(&segmax[(size_t)idx0[t] * 8 + h], fenc(alpha[i]));
}

__global__ void k3_exp(float* __restrict__ alpha, const int* __restrict__ idx0,
                       const unsigned* __restrict__ segmax,
                       float* __restrict__ segsum, int n) {
  int i = blockIdx.x * 256 + threadIdx.x;
  if (i >= n) return;
  int t = i >> 3, h = i & 7;
  size_t s = (size_t)idx0[t] * 8 + h;
  float ex = __expf(alpha[i] - fdec(segmax[s]));
  alpha[i] = ex;
  atomicAdd(&segsum[s], ex);
}

__global__ void k_off(const int* __restrict__ idx0, int* __restrict__ offs,
                      int E, int T) {
  int e = blockIdx.x * 256 + threadIdx.x;
  if (e > E) return;
  int lo = 0, hi = T;
  while (lo < hi) {
    int mid = (lo + hi) >> 1;
    if (idx0[mid] < e) lo = mid + 1; else hi = mid;
  }
  offs[e] = lo;
}

// =============== K4: attention-weighted gather/scatter (CSR, no atomics) =====
__global__ void __launch_bounds__(256)
k4_scatter(const int* __restrict__ offs, const int* __restrict__ tidx1,
           const int* __restrict__ inv_idx, const float* __restrict__ alpha,
           const float* __restrict__ segsum, const bf16_t* __restrict__ value,
           bf16_t* __restrict__ efea) {
  const int e = blockIdx.x;
  const int start = offs[e], end = offs[e + 1];
  const int f0 = threadIdx.x, f1 = threadIdx.x + 256;
  float a0 = 0.f, a1 = 0.f;
  if (start < end) {
    const int h0 = f0 >> 6, h1 = f1 >> 6;
    const float inv0 = fast_rcp(segsum[(size_t)e * 8 + h0]);
    const float inv1 = fast_rcp(segsum[(size_t)e * 8 + h1]);
    for (int t = start; t < end; ++t) {
      const int src = inv_idx[tidx1[t]];
      const bf16_t* vr = value + (size_t)src * 512;
      const float w0 = alpha[(size_t)t * 8 + h0] * inv0;
      const float w1 = alpha[(size_t)t * 8 + h1] * inv1;
      a0 += w0 * (float)vr[f0];
      a1 += w1 * (float)vr[f1];
    }
  }
  efea[(size_t)e * 512 + f0] = (bf16_t)a0;
  efea[(size_t)e * 512 + f1] = (bf16_t)a1;
}

// =============== K5: edge_out = edge_fea @ lin_W + b =========================
__global__ void __launch_bounds__(256)
k5_out(const bf16_t* __restrict__ efea, const bf16_t* __restrict__ pLin,
       const float* __restrict__ lb, float* __restrict__ out) {
  __shared__ alignas(16) bf16_t at[16 * LDB];
  const int e0 = blockIdx.x * 16, tid = threadIdx.x;
#if USE_TDM
  if (tid < 32) {
    // bf16 tile 16x512, pad 4 dwords every 256 dwords -> ld 520 bf16
    tdm_load_tile_2d(efea + (size_t)e0 * 512, (unsigned)(unsigned long long)(void*)at,
                     16, 512, 512, 1, 1, 7, 3);
    __builtin_amdgcn_s_wait_tensorcnt(0);
  }
#else
  for (int i = tid; i < 16 * 64; i += 256) {
    int r = i >> 6, c = i & 63;
    ((uint4*)(at + r * LDB))[c] = ((const uint4*)(efea + (size_t)(e0 + r) * 512))[c];
  }
#endif
  __syncthreads();
  const int w = tid >> 5, lane = tid & 31, m = lane & 15, hf = lane >> 4;
  v8f acc = {0.f, 0.f, 0.f, 0.f, 0.f, 0.f, 0.f, 0.f};
#pragma unroll
  for (int kc = 0; kc < 16; ++kc) {
    v16bf a = buildA_bf16(at, LDB, m, kc * 32, hf);
    v16bf b = load_bfrag(pLin, 8, kc, w, lane);
    acc = wmma_bf16(a, b, acc);
  }
  int n = w * 16 + m;
  float bv = lb[n];
#pragma unroll
  for (int i = 0; i < 8; ++i)
    out[(size_t)(e0 + i + hf * 8) * 128 + n] = acc[i] + bv;
}

// =============================================================================
extern "C" void kernel_launch(void* const* d_in, const int* in_sizes, int n_in,
                              void* d_out, int out_size, void* d_ws, size_t ws_size,
                              hipStream_t stream) {
  (void)n_in; (void)out_size;
  const float* edge_in = (const float*)d_in[0];
  const int*   inv_idx = (const int*)d_in[1];
  const float* edge_sh = (const float*)d_in[2];
  const float* elen    = (const float*)d_in[3];
  const int*   tidx0   = (const int*)d_in[6];
  const int*   tidx1   = (const int*)d_in[7];
  const float* temb2   = (const float*)d_in[8];
  const float* temb3   = (const float*)d_in[9];
  const float* W_tp    = (const float*)d_in[11];
  const float* rW1     = (const float*)d_in[12];
  const float* rb1     = (const float*)d_in[13];
  const float* rW2     = (const float*)d_in[14];
  const float* rb2     = (const float*)d_in[15];
  const float* aW1     = (const float*)d_in[16];
  const float* ab1     = (const float*)d_in[17];
  const float* ag1     = (const float*)d_in[18];
  const float* abe1    = (const float*)d_in[19];
  const float* aW2     = (const float*)d_in[20];
  const float* ab2     = (const float*)d_in[21];
  const float* ag2     = (const float*)d_in[22];
  const float* abe2    = (const float*)d_in[23];
  const float* aW3     = (const float*)d_in[24];
  const float* ab3     = (const float*)d_in[25];
  const float* linW    = (const float*)d_in[26];
  const float* linb    = (const float*)d_in[27];
  float* out = (float*)d_out;

  const int E = in_sizes[0] / 64;
  const int T = in_sizes[6];

  char* ws = (char*)d_ws;
  size_t off = 0;
  auto alloc = [&](size_t bytes) {
    size_t o = off; off = (off + bytes + 255) & ~(size_t)255; return o;
  };
  size_t o_pWtp  = alloc((size_t)1024 * 512 * 2);
  size_t o_pR1   = alloc((size_t)64 * 64 * 2);
  size_t o_pR2   = alloc((size_t)64 * 512 * 2);
  size_t o_pLin  = alloc((size_t)512 * 128 * 2);
  size_t o_pA1   = alloc((size_t)2 * 64 * 64 * 2);
  size_t o_pA2   = alloc((size_t)2 * 64 * 64 * 2);
  size_t o_pA3   = alloc((size_t)2 * 64 * 16 * 2);
  size_t o_val   = alloc((size_t)E * 512 * 2);
  size_t o_alpha = alloc((size_t)T * 8 * 4);
  size_t o_smax  = alloc((size_t)E * 8 * 4);
  size_t o_ssum  = alloc((size_t)E * 8 * 4);
  size_t o_offs  = alloc((size_t)(E + 1) * 4);
  size_t o_efea  = alloc((size_t)E * 512 * 2);
  if (off > ws_size) return;  // workspace too small; nothing safe to do

  bf16_t* pWtp = (bf16_t*)(ws + o_pWtp);
  bf16_t* pR1  = (bf16_t*)(ws + o_pR1);
  bf16_t* pR2  = (bf16_t*)(ws + o_pR2);
  bf16_t* pLin = (bf16_t*)(ws + o_pLin);
  bf16_t* pA1  = (bf16_t*)(ws + o_pA1);
  bf16_t* pA2  = (bf16_t*)(ws + o_pA2);
  bf16_t* pA3  = (bf16_t*)(ws + o_pA3);
  bf16_t* val  = (bf16_t*)(ws + o_val);
  float*  alphab = (float*)(ws + o_alpha);
  unsigned* smax = (unsigned*)(ws + o_smax);
  float*  ssum = (float*)(ws + o_ssum);
  int*    offs = (int*)(ws + o_offs);
  bf16_t* efea = (bf16_t*)(ws + o_efea);

  (void)hipMemsetAsync(smax, 0, (size_t)E * 8 * 4, stream);  // 0 == encoded -inf
  (void)hipMemsetAsync(ssum, 0, (size_t)E * 8 * 4, stream);

  auto packN = [](int K, int N) { return (K >> 5) * (N >> 4) * 32; };
  pack_b_kernel<<<(packN(1024, 512) + 255) / 256, 256, 0, stream>>>(W_tp, pWtp, 1024, 512, 512);
  pack_b_kernel<<<1, 256, 0, stream>>>(rW1, pR1, 64, 64, 64);
  pack_b_kernel<<<(packN(64, 512) + 255) / 256, 256, 0, stream>>>(rW2, pR2, 64, 512, 512);
  pack_b_kernel<<<(packN(512, 128) + 255) / 256, 256, 0, stream>>>(linW, pLin, 512, 128, 128);
  for (int br = 0; br < 2; ++br) {
    pack_b_kernel<<<1, 256, 0, stream>>>(aW1 + br * 64 * 64, pA1 + br * 4096, 64, 64, 64);
    pack_b_kernel<<<1, 256, 0, stream>>>(aW2 + br * 64 * 64, pA2 + br * 4096, 64, 64, 64);
    pack_b_kernel<<<1, 256, 0, stream>>>(aW3 + br * 64 * 8,  pA3 + br * 1024, 64, 16, 8);
  }

  k1_value<<<E / 16, 256, 0, stream>>>(edge_in, edge_sh, elen, pWtp, pR1, pR2,
                                       rb1, rb2, val);
  k2_alpha<<<T / 16, 256, 0, stream>>>(temb2, temb3, pA1, pA2, pA3, ab1, ag1,
                                       abe1, ab2, ag2, abe2, ab3, alphab);
  int n8 = T * 8;
  k3_max<<<(n8 + 255) / 256, 256, 0, stream>>>(alphab, tidx0, smax, n8);
  k3_exp<<<(n8 + 255) / 256, 256, 0, stream>>>(alphab, tidx0, smax, ssum, n8);
  k_off<<<(E + 256) / 256, 256, 0, stream>>>(tidx0, offs, E, T);
  k4_scatter<<<E, 256, 0, stream>>>(offs, tidx1, inv_idx, alphab, ssum, val, efea);
  k5_out<<<E / 16, 256, 0, stream>>>(efea, pLin, linb, out);
}
